// CopyGenerator_18760417148948
// MI455X (gfx1250) — compile-verified, matrix operand order checked by reading, not run on hardware
//
#include <hip/hip_runtime.h>
#include <hip/hip_bf16.h>

// ---------------------------------------------------------------------------
// CopyGenerator for MI455X (gfx1250, wave32, WMMA + Tensor Data Mover).
//   B=16, T=128, S=512, C=512, V=50000, D=1024; N=B*T=2048; out=[2048,50512] f32
// Main GEMM (2048x50000x1024) on v_wmma_f32_16x16x32_bf16; tiles staged by TDM
// (tensor_load_to_lds, TENSORcnt, double-buffered LDS) after one-time f32->bf16
// conversion of W/hidden (bf16 W = 102 MB -> resident in 192 MB L2).
// ---------------------------------------------------------------------------

typedef __bf16 v16bf __attribute__((ext_vector_type(16)));
typedef __bf16 v8bf  __attribute__((ext_vector_type(8)));
typedef float  v8f   __attribute__((ext_vector_type(8)));

typedef unsigned int tdm_v4u __attribute__((ext_vector_type(4)));
typedef int          tdm_v4i __attribute__((ext_vector_type(4)));
typedef int          tdm_v8i __attribute__((ext_vector_type(8)));

union FragBF { v16bf v; v8bf h[2]; };

#define BM 128
#define BN 128
#define BK 32
#define LDSS 40   // padded bf16 row stride: 80 bytes (16B aligned; TDM pad 4dw/16dw)

// ---------------------------------------------------------------------------
// TDM: DMA a [rows x 32 bf16] tile (row stride = stride_elems) into LDS with
// 16B padding after each 64B row (-> LDSS row pitch). Rows past rows_avail are
// zero-filled by TDM OOB handling (covers the N-edge tile of the big GEMM).
// D# layout per CDNA5 ISA 8.3/8.4 (group0: count/lds/global/type,
// group1: data_size=2B, pad_interval=16dw(f=3), pad_amount=4dw(f=3),
// tensor_dim0=32, tensor_dim1=rows_avail, tile=32x128, dim0_stride).
// ---------------------------------------------------------------------------
__device__ __forceinline__
void tdm_load_tile(unsigned lds_addr, const __bf16* gptr, int rows_avail,
                   int stride_elems, int tile_rows)
{
    unsigned long long ga = (unsigned long long)(size_t)gptr;

    tdm_v4u g0;
    g0[0] = 1u;                                            // count=1, user desc
    g0[1] = lds_addr;                                      // LDS byte address
    g0[2] = (unsigned)(ga & 0xffffffffu);                  // global_addr[31:0]
    g0[3] = (unsigned)((ga >> 32) & 0x01ffffffu)           // global_addr[56:32]
          | 0x80000000u;                                   // type=2 (image)

    unsigned rows = (unsigned)rows_avail;
    tdm_v8i g1;
    g1[0] = (int)((1u << 16)            // data_size = 2 bytes
                | (1u << 20)            // pad_enable
                | (3u << 22)            // pad_interval: 16 DWORDs (64B row)
                | (3u << 25));          // pad_amount: 4 DWORDs (16B)
    g1[1] = (int)(32u << 16);                           // tensor_dim0[15:0]=32
    g1[2] = (int)((rows & 0xffffu) << 16);              // dim0[31:16]=0 | tensor_dim1[15:0]
    g1[3] = (int)((rows >> 16) | (32u << 16));          // tensor_dim1[31:16] | tile_dim0=32
    g1[4] = (int)((unsigned)tile_rows);                 // tile_dim1 | tile_dim2=0
    g1[5] = (int)((unsigned)stride_elems);              // tensor_dim0_stride[31:0]
    g1[6] = 0;                                          // stride[47:32] | dim1_stride lo
    g1[7] = 0;

    tdm_v4i z4 = {0, 0, 0, 0};
#if __clang_major__ >= 23
    tdm_v8i z8 = {0, 0, 0, 0, 0, 0, 0, 0};
    __builtin_amdgcn_tensor_load_to_lds(g0, g1, z4, z4, z8, 0);
#else
    __builtin_amdgcn_tensor_load_to_lds(g0, g1, z4, z4, 0);
#endif
}

// ---------------------------------------------------------------------------
// Shared tiled GEMM: C[z] = A[z] (MxK bf16 row-major) * B[z]^T (B stored NxK
// bf16 row-major) + bias.  8 waves/block, each wave owns a 16x128 strip:
// 8 x v_wmma_f32_16x16x32_bf16 per BK step. TDM double-buffered staging.
// ---------------------------------------------------------------------------
__global__ __launch_bounds__(256)
void gemm_bf16_wmma(const __bf16* __restrict__ A, const __bf16* __restrict__ B,
                    float* __restrict__ C, const float* __restrict__ bias,
                    int M, int N, int K, int lda, int ldb, int ldc,
                    long long Abs, long long Bbs, long long Cbs)
{
    __shared__ __bf16 lA[2][BM * LDSS];
    __shared__ __bf16 lB[2][BN * LDSS];

    const int tid  = threadIdx.x;
    const int lane = tid & 31;
    const int w    = tid >> 5;       // wave id 0..7
    const int lh   = lane >> 4;      // half-wave select (K-half for A/B frags)
    const int r16  = lane & 15;

    const __bf16* Ag = A + (size_t)blockIdx.z * (size_t)Abs;
    const __bf16* Bg = B + (size_t)blockIdx.z * (size_t)Bbs;
    float*        Cg = C + (size_t)blockIdx.z * (size_t)Cbs;

    const int tileM = blockIdx.x * BM;
    const int tileN = blockIdx.y * BN;
    const int m0    = w << 4;        // wave's 16-row sub-tile inside block tile

    const __bf16* Atile = Ag + (size_t)tileM * lda;   // rows tileM.., all of K
    const __bf16* Btile = Bg + (size_t)tileN * ldb;
    const int     rowsB = N - tileN;                  // TDM zero-fills past edge

    const unsigned ldsA[2] = { (unsigned)(size_t)&lA[0][0], (unsigned)(size_t)&lA[1][0] };
    const unsigned ldsB[2] = { (unsigned)(size_t)&lB[0][0], (unsigned)(size_t)&lB[1][0] };

    if (w == 0) {                                     // prime buffer 0
        tdm_load_tile(ldsA[0], Atile, BM, lda, BM);
        tdm_load_tile(ldsB[0], Btile, rowsB, ldb, BN);
    }

    v8f acc[8] = {};

    for (int k0 = 0, it = 0; k0 < K; k0 += BK, ++it) {
        const int bufc = it & 1;
        if (w == 0) __builtin_amdgcn_s_wait_tensorcnt(0);   // current tiles landed
        __syncthreads();                                    // publish LDS tiles
        if (w == 0 && (k0 + BK) < K) {                      // stream next tiles
            tdm_load_tile(ldsA[bufc ^ 1], Atile + (k0 + BK), BM, lda, BM);
            tdm_load_tile(ldsB[bufc ^ 1], Btile + (k0 + BK), rowsB, ldb, BN);
        }

        const __bf16* sA = lA[bufc];
        const __bf16* sB = lB[bufc];

        // A frag 16x32: lanes 0-15 hold K{0..7,16..23}, lanes 16-31 K{8..15,24..31}
        FragBF fa;
        fa.h[0] = *(const v8bf*)&sA[(m0 + r16) * LDSS + (lh << 3)];
        fa.h[1] = *(const v8bf*)&sA[(m0 + r16) * LDSS + 16 + (lh << 3)];

        FragBF fb[8];
        #pragma unroll
        for (int j = 0; j < 8; ++j) {
            const int nr = (j << 4) + r16;
            fb[j].h[0] = *(const v8bf*)&sB[nr * LDSS + (lh << 3)];
            fb[j].h[1] = *(const v8bf*)&sB[nr * LDSS + 16 + (lh << 3)];
        }
        #pragma unroll
        for (int j = 0; j < 8; ++j)
            acc[j] = __builtin_amdgcn_wmma_f32_16x16x32_bf16(
                         false, fa.v, false, fb[j].v, (short)0, acc[j], false, false);

        __syncthreads();   // everyone done with bufc before it is DMA-overwritten
    }

    // ---- epilogue: C/D layout: lane<16 -> M=r (VGPR r), N=lane;
    //                lane>=16 -> M=8+r, N=lane-16
    const int mBase = tileM + (w << 4) + (lh << 3);
    #pragma unroll
    for (int j = 0; j < 8; ++j) {
        const int n = tileN + (j << 4) + r16;
        if (n < N) {
            const float bv = bias ? bias[n] : 0.0f;
            float* cp = Cg + (size_t)mBase * ldc + n;
            #pragma unroll
            for (int r = 0; r < 8; ++r)
                cp[(size_t)r * ldc] = acc[j][r] + bv;
        }
    }
}

// ---------------------------------------------------------------------------
// f32 -> bf16 flat conversion, 4 elems/thread via float4 loads
// ---------------------------------------------------------------------------
__global__ __launch_bounds__(256)
void cvt_f32_bf16(const float* __restrict__ in, __bf16* __restrict__ out, long long n)
{
    long long i = ((long long)blockIdx.x * 256 + threadIdx.x) * 4;
    if (i + 3 < n) {
        float4 v = *(const float4*)(in + i);
        out[i + 0] = (__bf16)v.x;
        out[i + 1] = (__bf16)v.y;
        out[i + 2] = (__bf16)v.z;
        out[i + 3] = (__bf16)v.w;
    } else {
        for (; i < n; ++i) out[i] = (__bf16)in[i];
    }
}

// ---------------------------------------------------------------------------
// p_copy[row] = sigmoid(hidden[row,:] . w_copy + b_copy)   (one block per row)
// ---------------------------------------------------------------------------
__global__ __launch_bounds__(256)
void pcopy_kernel(const float* __restrict__ hidden, const float* __restrict__ wcopy,
                  const float* __restrict__ bcopy, float* __restrict__ pcopy, int D)
{
    __shared__ float red[256];
    const int row = blockIdx.x;
    const float* h = hidden + (size_t)row * D;
    float s = 0.0f;
    for (int i = threadIdx.x; i < D; i += 256) s += h[i] * wcopy[i];
    red[threadIdx.x] = s;
    __syncthreads();
    for (int st = 128; st > 0; st >>= 1) {
        if (threadIdx.x < st) red[threadIdx.x] += red[threadIdx.x + st];
        __syncthreads();
    }
    if (threadIdx.x == 0) {
        float x = red[0] + bcopy[0];
        pcopy[row] = 1.0f / (1.0f + __expf(-x));
    }
}

// ---------------------------------------------------------------------------
// mul_attn = (attn * p_copy[row]) -> bf16
// ---------------------------------------------------------------------------
__global__ __launch_bounds__(256)
void mulattn_kernel(const float* __restrict__ attn, const float* __restrict__ pcopy,
                    __bf16* __restrict__ out, int S, long long total)
{
    long long idx = (long long)blockIdx.x * 256 + threadIdx.x;
    if (idx < total) {
        int r = (int)(idx / S);
        out[idx] = (__bf16)(attn[idx] * pcopy[r]);
    }
}

// ---------------------------------------------------------------------------
// srcT[b][c][s] = (bf16) src_map[b][s][c]   (32x32 LDS tile transpose)
// ---------------------------------------------------------------------------
__global__ __launch_bounds__(256)
void transpose_cvt(const float* __restrict__ src, __bf16* __restrict__ dst, int S, int C)
{
    __shared__ float tile[32][33];
    const int b  = blockIdx.z;
    const int c0 = blockIdx.x * 32;
    const int s0 = blockIdx.y * 32;
    const float* sp = src + (size_t)b * S * C;
    __bf16*      dp = dst + (size_t)b * S * C;
    const int tx  = threadIdx.x & 31;
    const int ty0 = threadIdx.x >> 5;
    for (int ty = ty0; ty < 32; ty += 8)
        tile[ty][tx] = sp[(size_t)(s0 + ty) * C + c0 + tx];
    __syncthreads();
    for (int ty = ty0; ty < 32; ty += 8)
        dp[(size_t)(c0 + ty) * S + s0 + tx] = (__bf16)tile[tx][ty];
}

// ---------------------------------------------------------------------------
// In-place masked softmax per row, scaled by (1 - p_copy). pad_idx read on
// device (it lives in device memory; no host copies under graph capture).
// ---------------------------------------------------------------------------
__global__ __launch_bounds__(256)
void softmax_rows(float* __restrict__ out, const float* __restrict__ pcopy,
                  const int* __restrict__ pad_ptr, int V, int ldc)
{
    __shared__ float red[256];
    const int row = blockIdx.x;
    const int pad = pad_ptr[0];
    float* base = out + (size_t)row * ldc;
    const int tid = threadIdx.x;

    float m = -3.402823466e38f;
    for (int i = tid; i < V; i += 256)
        if (i != pad) m = fmaxf(m, base[i]);
    red[tid] = m; __syncthreads();
    for (int st = 128; st > 0; st >>= 1) {
        if (tid < st) red[tid] = fmaxf(red[tid], red[tid + st]);
        __syncthreads();
    }
    m = red[0];
    __syncthreads();

    float s = 0.0f;
    for (int i = tid; i < V; i += 256)
        if (i != pad) s += __expf(base[i] - m);
    red[tid] = s; __syncthreads();
    for (int st = 128; st > 0; st >>= 1) {
        if (tid < st) red[tid] += red[tid + st];
        __syncthreads();
    }
    s = red[0];

    const float scale = (1.0f - pcopy[row]) / s;
    for (int i = tid; i < V; i += 256)
        base[i] = (i == pad) ? 0.0f : __expf(base[i] - m) * scale;
}

// ---------------------------------------------------------------------------
extern "C" void kernel_launch(void* const* d_in, const int* in_sizes, int n_in,
                              void* d_out, int out_size, void* d_ws, size_t ws_size,
                              hipStream_t stream)
{
    (void)in_sizes; (void)n_in; (void)out_size; (void)ws_size;

    constexpr int Bb = 16, T = 128, S = 512, Cc = 512, V = 50000, D = 1024;
    constexpr int NR  = Bb * T;       // 2048 rows
    constexpr int LDC = V + Cc;       // 50512

    const float* hidden  = (const float*)d_in[0];
    const float* attn    = (const float*)d_in[1];
    const float* src_map = (const float*)d_in[2];
    const float* W       = (const float*)d_in[3];
    const float* bvec    = (const float*)d_in[4];
    const float* w_copy  = (const float*)d_in[5];
    const float* b_copy  = (const float*)d_in[6];
    const int*   pad_ptr = (const int*)d_in[7];
    float*       out     = (float*)d_out;

    // workspace layout (all offsets 256B aligned), ~112 MB total
    char* ws = (char*)d_ws;
    size_t off = 0;
    __bf16* Abf   = (__bf16*)(ws + off); off += (size_t)NR * D * 2;     //  4.19 MB
    __bf16* Wbf   = (__bf16*)(ws + off); off += (size_t)V  * D * 2;     // 102.4 MB
    float*  pcp   = (float*) (ws + off); off += (size_t)NR * 4;         //  8 KB
    __bf16* MAbf  = (__bf16*)(ws + off); off += (size_t)NR * S * 2;     //  2.1 MB
    __bf16* SMTbf = (__bf16*)(ws + off);                                //  8.4 MB

    // 1) bf16 conversions (W in bf16 = 102 MB -> resident in 192 MB L2)
    {
        long long nA = (long long)NR * D;
        long long nW = (long long)V * D;
        cvt_f32_bf16<<<(unsigned)((nA / 4 + 255) / 256), 256, 0, stream>>>(hidden, Abf, nA);
        cvt_f32_bf16<<<(unsigned)((nW / 4 + 255) / 256), 256, 0, stream>>>(W, Wbf, nW);
    }

    // 2) copy gate
    pcopy_kernel<<<NR, 256, 0, stream>>>(hidden, w_copy, b_copy, pcp, D);

    // 3) mul_attn = attn * p_copy  (bf16)
    {
        long long tot = (long long)NR * S;
        mulattn_kernel<<<(unsigned)((tot + 255) / 256), 256, 0, stream>>>(attn, pcp, MAbf, S, tot);
    }

    // 4) src_map transposed to [b][c][s] bf16 so copy-GEMM B is NxK row-major
    transpose_cvt<<<dim3(Cc / 32, S / 32, Bb), 256, 0, stream>>>(src_map, SMTbf, S, Cc);

    // 5) main logits GEMM: out[:, 0:V] = hidden @ W^T + b
    gemm_bf16_wmma<<<dim3(NR / BM, (V + BN - 1) / BN, 1), 256, 0, stream>>>(
        Abf, Wbf, out, bvec, NR, V, D, D, D, LDC, 0, 0, 0);

    // 6) masked softmax * (1 - p_copy), in place
    softmax_rows<<<NR, 256, 0, stream>>>(out, pcp, pad_ptr, V, LDC);

    // 7) copy_prob batched GEMM: out[:, V:V+C] = mul_attn @ src_map (per batch)
    gemm_bf16_wmma<<<dim3(1, Cc / BN, Bb), 256, 0, stream>>>(
        MAbf, SMTbf, out + V, nullptr, T, Cc, S, S, S, LDC,
        (long long)T * S, (long long)S * Cc, (long long)T * LDC);
}
